// CasparLayer_14259291423183
// MI455X (gfx1250) — matrix operand order repaired and checked by reading, not compiled, not run on hardware
//
#include <hip/hip_runtime.h>
#include <hip/hip_bf16.h>
#include <math.h>
#include <stdint.h>

#define VOCAB 41
#define EMB   512
#define HID   1024
#define H3    3072
#define NB    128
#define NT    256
#define NBLK  64          // blocks in persistent GRU kernel

typedef __bf16 bf16;
typedef __attribute__((ext_vector_type(16))) __bf16 v16bf;
typedef __attribute__((ext_vector_type(8)))  __bf16 v8bf;
typedef __attribute__((ext_vector_type(8)))  float  v8f;

// gcc-style int4 to exactly match the async-LDS builtin's parameter type
typedef int v4i __attribute__((vector_size(16)));
typedef __attribute__((address_space(1))) v4i as1_v4i;
typedef __attribute__((address_space(3))) v4i as3_v4i;

#if defined(__has_builtin)
#if __has_builtin(__builtin_amdgcn_global_load_async_to_lds_b128) && \
    __has_builtin(__builtin_amdgcn_s_wait_asynccnt)
#define HAVE_ASYNC_LDS 1
#endif
#endif
#ifndef HAVE_ASYNC_LDS
#define HAVE_ASYNC_LDS 0
#endif

union V16U { v16bf v; v8bf h[2]; };
union BFU  { unsigned short u; bf16 b; };

__device__ __forceinline__ bf16 to_bf16(float f) {
  unsigned u = __float_as_uint(f);
  u = (u + 0x7FFFu + ((u >> 16) & 1u)) >> 16;
  BFU x; x.u = (unsigned short)u; return x.b;
}

// ---- one-time prep kernels -------------------------------------------------

// ptab[v][c] = sum_k emb[v][k] * gru_kernel[k][c] + gru_bias[0][c]
__global__ void ptab_kernel(const float* __restrict__ emb,
                            const float* __restrict__ gk,
                            const float* __restrict__ gbias0,
                            float* __restrict__ ptab) {
  int c = blockIdx.x * blockDim.x + threadIdx.x;   // 0..3071
  int v = blockIdx.y;                              // 0..40
  float acc = gbias0[c];
  const float* e = emb + v * EMB;
  for (int k = 0; k < EMB; ++k)
    acc = fmaf(e[k], gk[(size_t)k * H3 + c], acc);
  ptab[(size_t)v * H3 + c] = acc;
}

// recT[c][k] = bf16(rec_kernel[k][c])  (transpose + convert, [3072,1024])
__global__ void cvt_rec_kernel(const float* __restrict__ rec,
                               bf16* __restrict__ recT) {
  int i = blockIdx.x * blockDim.x + threadIdx.x;
  int c = i >> 10, k = i & 1023;
  recT[i] = to_bf16(rec[(size_t)k * H3 + c]);
}

// dwT[n][k] = bf16(dense_w[k][n]) for n<41, else 0   ([48,1024])
__global__ void cvt_dense_kernel(const float* __restrict__ dw,
                                 bf16* __restrict__ dwT) {
  int i = blockIdx.x * blockDim.x + threadIdx.x;
  int n = i >> 10, k = i & 1023;
  dwT[i] = (n < VOCAB) ? to_bf16(dw[(size_t)k * VOCAB + n]) : to_bf16(0.0f);
}

__global__ void zero_h_kernel(float* __restrict__ hf, bf16* __restrict__ hb,
                              unsigned* __restrict__ cnt) {
  int i = blockIdx.x * blockDim.x + threadIdx.x;   // 0..128*1024-1
  hf[i] = 0.0f;
  hb[i] = to_bf16(0.0f);
  if (i == 0) *cnt = 0u;
}

// ---- WMMA tile loaders (16-bit A/B, contiguous k order per lane half) ---------------------

// A 16x32 (MxK), row-major source, ld = leading dim:
// lanes 0-15: M=lane, K=k0..k0+15; lanes 16-31: M=lane-16, K=k0+16..k0+31
__device__ __forceinline__ v16bf load_tileA(const bf16* __restrict__ base,
                                            int ld, int row0, int k0, int lane) {
  int r   = row0 + (lane & 15);
  int klo = k0 + ((lane >> 4) << 4);
  const bf16* p = base + (size_t)r * ld + klo;
  V16U u;
  u.h[0] = *(const v8bf*)(p);
  u.h[1] = *(const v8bf*)(p + 8);
  return u.v;
}

// B 32x16 (KxN) from column-major source ([N,K] row-major):
// lanes 0-15: N=lane, K=k0..k0+15; lanes 16-31: N=lane-16, K=k0+16..k0+31
__device__ __forceinline__ v16bf load_tileB(const bf16* __restrict__ baseT,
                                            int ld, int col0, int k0, int lane) {
  int c  = col0 + (lane & 15);
  int kk = k0 + ((lane >> 4) << 4);
  const bf16* p = baseT + (size_t)c * ld + kk;
  V16U u;
  u.h[0] = *(const v8bf*)(p);
  u.h[1] = *(const v8bf*)(p + 8);
  return u.v;
}

// Same as load_tileB but from the block's LDS-resident weight slice [48][HID]
__device__ __forceinline__ v16bf load_tileB_lds(const bf16* Bs, int gate,
                                                int k0, int lane) {
  int lcol = (gate << 4) + (lane & 15);
  int kk   = k0 + ((lane >> 4) << 4);
  const bf16* p = Bs + lcol * HID + kk;
  V16U u;
  u.h[0] = *(const v8bf*)(p);       // ds_load_b128
  u.h[1] = *(const v8bf*)(p + 8);   // ds_load_b128
  return u.v;
}

// ---- device-wide barrier for the persistent GRU kernel ---------------------

__device__ __forceinline__ void grid_sync(unsigned* cnt, unsigned expected) {
  __threadfence();                  // every wave drains its global stores
  __syncthreads();
  if (threadIdx.x == 0) {
    __hip_atomic_fetch_add(cnt, 1u, __ATOMIC_RELEASE, __HIP_MEMORY_SCOPE_AGENT);
    while (__hip_atomic_load(cnt, __ATOMIC_ACQUIRE, __HIP_MEMORY_SCOPE_AGENT)
           < expected)
      __builtin_amdgcn_s_sleep(2);
  }
  __syncthreads();
}

// ---- GRU gating ------------------------------------------------------------

__device__ __forceinline__ void gate_and_store(
    v8f accZ, v8f accR, v8f accH, int mbase, int mhi, int j, int t,
    float bz_, float br_, float bh_,
    const int* __restrict__ x, const unsigned char* __restrict__ mask,
    const float* __restrict__ ptab, const float* __restrict__ hprev_f,
    float* __restrict__ hnext_f, bf16* __restrict__ hnext_b,
    bf16* __restrict__ hseq) {
#pragma unroll
  for (int g = 0; g < 8; ++g) {
    int m = mbase + mhi + g;                       // batch row
    int tok = x[m * NT + t];
    const float* pt = ptab + (size_t)tok * H3;
    float xz = pt[j], xr = pt[HID + j], xh = pt[2 * HID + j];
    float hp = hprev_f[(size_t)m * HID + j];
    float z  = 1.0f / (1.0f + __expf(-(xz + accZ[g] + bz_)));
    float r  = 1.0f / (1.0f + __expf(-(xr + accR[g] + br_)));
    float hh = tanhf(xh + r * (accH[g] + bh_));
    float hn = z * hp + (1.0f - z) * hh;
    float ho = mask[m * NT + t] ? hn : hp;         // keras masked-step carry
    hnext_f[(size_t)m * HID + j] = ho;
    bf16 hob = to_bf16(ho);
    hnext_b[(size_t)m * HID + j] = hob;
    hseq[((size_t)m * NT + t) * HID + j] = hob;
  }
}

// ---- persistent GRU: all 256 timesteps in one launch -----------------------
// 64 blocks x 128 threads. Block b owns hidden units [16b, 16b+16) and keeps
// its 48 recurrent-weight columns (z/r/h) resident in LDS for the whole scan.

__global__ void __launch_bounds__(128)
gru_persistent_kernel(const int* __restrict__ x,
                      const unsigned char* __restrict__ mask,
                      const float* __restrict__ ptab,
                      const bf16* __restrict__ recT,
                      const float* __restrict__ rbias,
                      float* __restrict__ hf0, bf16* __restrict__ hb0,
                      float* __restrict__ hf1, bf16* __restrict__ hb1,
                      bf16* __restrict__ hseq,
                      unsigned* __restrict__ barrier_cnt) {
  extern __shared__ char smem[];
  bf16* Bs = (bf16*)smem;                 // [48][HID] bf16 = 96 KB

  int lane = threadIdx.x & 31;
  int wave = threadIdx.x >> 5;
  int j0   = blockIdx.x << 4;             // 16 hidden units per block
  int m0   = wave << 5;                   // 32 batch rows (2 M-tiles) per wave

  // ---- stage recurrent weights into LDS once (time-invariant) ----
  for (int c = threadIdx.x; c < 48 * (HID / 8); c += 128) {
    int lcol = c >> 7;                    // 0..47
    int koff = (c & 127) << 3;            // element offset, 16B chunks
    int gate = lcol >> 4, col = lcol & 15;
    const bf16* src = recT + (size_t)(gate * HID + j0 + col) * HID + koff;
    bf16* dst = Bs + lcol * HID + koff;
#if HAVE_ASYNC_LDS
    __builtin_amdgcn_global_load_async_to_lds_b128(
        (as1_v4i*)(uintptr_t)src,
        (as3_v4i*)(unsigned)(uintptr_t)dst, 0, 0);
#else
    *(v8bf*)dst = *(const v8bf*)src;
#endif
  }
#if HAVE_ASYNC_LDS
  __builtin_amdgcn_s_wait_asynccnt(0);
#endif
  __syncthreads();

  int n   = lane & 15;
  int j   = j0 + n;
  int mhi = (lane >> 4) << 3;
  float bz_ = rbias[j], br_ = rbias[HID + j], bh_ = rbias[2 * HID + j];

  for (int t = 0; t < NT; ++t) {
    const float* hpf = (t & 1) ? hf1 : hf0;
    const bf16*  hpb = (t & 1) ? hb1 : hb0;
    float*       hnf = (t & 1) ? hf0 : hf1;
    bf16*        hnb = (t & 1) ? hb0 : hb1;

    v8f accZ0 = {}, accR0 = {}, accH0 = {};
    v8f accZ1 = {}, accR1 = {}, accH1 = {};

    for (int k = 0; k < HID; k += 32) {
      if (k + 32 < HID)   // keep next h-state K-slice moving toward L0
        __builtin_prefetch(hpb + (size_t)(m0 + (lane & 15)) * HID + k + 32, 0, 0);
      v16bf a0 = load_tileA(hpb, HID, m0,      k, lane);
      v16bf a1 = load_tileA(hpb, HID, m0 + 16, k, lane);
      v16bf bz = load_tileB_lds(Bs, 0, k, lane);
      v16bf br = load_tileB_lds(Bs, 1, k, lane);
      v16bf bh = load_tileB_lds(Bs, 2, k, lane);
      accZ0 = __builtin_amdgcn_wmma_f32_16x16x32_bf16(false, a0, false, bz, (short)0, accZ0, false, false);
      accR0 = __builtin_amdgcn_wmma_f32_16x16x32_bf16(false, a0, false, br, (short)0, accR0, false, false);
      accH0 = __builtin_amdgcn_wmma_f32_16x16x32_bf16(false, a0, false, bh, (short)0, accH0, false, false);
      accZ1 = __builtin_amdgcn_wmma_f32_16x16x32_bf16(false, a1, false, bz, (short)0, accZ1, false, false);
      accR1 = __builtin_amdgcn_wmma_f32_16x16x32_bf16(false, a1, false, br, (short)0, accR1, false, false);
      accH1 = __builtin_amdgcn_wmma_f32_16x16x32_bf16(false, a1, false, bh, (short)0, accH1, false, false);
    }

    gate_and_store(accZ0, accR0, accH0, m0,      mhi, j, t, bz_, br_, bh_,
                   x, mask, ptab, hpf, hnf, hnb, hseq);
    gate_and_store(accZ1, accR1, accH1, m0 + 16, mhi, j, t, bz_, br_, bh_,
                   x, mask, ptab, hpf, hnf, hnb, hseq);

    grid_sync(barrier_cnt, (unsigned)NBLK * (unsigned)(t + 1));
  }
}

// ---- final dense: [32768,1024] x [1024,48] -> logits [32768,41] ------------

__global__ void __launch_bounds__(128)
dense_kernel(const bf16* __restrict__ hseq, const bf16* __restrict__ dwT,
             const float* __restrict__ db, float* __restrict__ out) {
  int lane = threadIdx.x & 31;
  int wave = threadIdx.x >> 5;
  int m0 = (blockIdx.x * 4 + wave) << 4;   // grid 512 -> covers 32768 rows

  v8f acc0 = {}, acc1 = {}, acc2 = {};
  for (int k = 0; k < HID; k += 32) {
    v16bf a  = load_tileA(hseq, HID, m0, k, lane);
    v16bf b0 = load_tileB(dwT, HID,  0, k, lane);
    v16bf b1 = load_tileB(dwT, HID, 16, k, lane);
    v16bf b2 = load_tileB(dwT, HID, 32, k, lane);
    acc0 = __builtin_amdgcn_wmma_f32_16x16x32_bf16(false, a, false, b0, (short)0, acc0, false, false);
    acc1 = __builtin_amdgcn_wmma_f32_16x16x32_bf16(false, a, false, b1, (short)0, acc1, false, false);
    acc2 = __builtin_amdgcn_wmma_f32_16x16x32_bf16(false, a, false, b2, (short)0, acc2, false, false);
  }

  int n = lane & 15;
  int mhi = (lane >> 4) << 3;
#pragma unroll
  for (int g = 0; g < 8; ++g) {
    int m = m0 + mhi + g;
    out[(size_t)m * VOCAB + n]      = acc0[g] + db[n];         // n  <= 15
    out[(size_t)m * VOCAB + n + 16] = acc1[g] + db[n + 16];    // <= 31
    int n2 = n + 32;
    if (n2 < VOCAB) out[(size_t)m * VOCAB + n2] = acc2[g] + db[n2];
  }
}

// ---- launcher --------------------------------------------------------------

extern "C" void kernel_launch(void* const* d_in, const int* in_sizes, int n_in,
                              void* d_out, int out_size, void* d_ws, size_t ws_size,
                              hipStream_t stream) {
  const int*            x    = (const int*)d_in[0];
  const unsigned char*  mask = (const unsigned char*)d_in[1];
  const float*          emb  = (const float*)d_in[2];
  const float*          gk   = (const float*)d_in[3];
  const float*          grk  = (const float*)d_in[4];
  const float*          gb   = (const float*)d_in[5];
  const float*          dw   = (const float*)d_in[6];
  const float*          db   = (const float*)d_in[7];
  float* out = (float*)d_out;

  char* ws = (char*)d_ws;                       // ~75.6 MB total
  float*    ptab = (float*)(ws + 0);            // 41*3072 f32     (503808 B)
  bf16*     recT = (bf16*)(ws + 503808);        // 3072*1024 bf16  (6291456 B)
  bf16*     dwT  = (bf16*)(ws + 6795264);       // 48*1024 bf16    (98304 B)
  float*    hf   = (float*)(ws + 6893568);      // 2 x 128*1024 f32
  bf16*     hb   = (bf16*)(ws + 7942144);       // 2 x 128*1024 bf16
  bf16*     hseq = (bf16*)(ws + 8466432);       // 128*256*1024 bf16 (64 MiB)
  unsigned* cnt  = (unsigned*)(ws + 75575296);  // barrier counter

  ptab_kernel<<<dim3(H3 / 256, VOCAB), 256, 0, stream>>>(emb, gk, gb, ptab);
  cvt_rec_kernel<<<(H3 * HID) / 256, 256, 0, stream>>>(grk, recT);
  cvt_dense_kernel<<<(48 * 1024) / 256, 256, 0, stream>>>(dw, dwT);
  zero_h_kernel<<<(NB * HID) / 256, 256, 0, stream>>>(hf, hb, cnt);

  const int HS = NB * HID;
  const int ldsBytes = 48 * HID * (int)sizeof(bf16);   // 96 KB
  (void)hipFuncSetAttribute((const void*)gru_persistent_kernel,
                            hipFuncAttributeMaxDynamicSharedMemorySize, ldsBytes);
  gru_persistent_kernel<<<NBLK, 128, ldsBytes, stream>>>(
      x, mask, ptab, recT, gb + H3,
      hf, hb, hf + (size_t)HS, hb + (size_t)HS, hseq, cnt);

  dense_kernel<<<512, 128, 0, stream>>>(hseq, dwT, db, out);
}
